// Fast_dLLM_QwenAttention_86337432584494
// MI455X (gfx1250) — matrix operand order repaired and checked by reading, not compiled, not run on hardware
//
#include <hip/hip_runtime.h>
#include <hip/hip_bf16.h>

typedef __bf16 bf16;
typedef __attribute__((ext_vector_type(16))) __bf16 v16bf;
typedef __attribute__((ext_vector_type(8)))  __bf16 v8bf;
typedef __attribute__((ext_vector_type(8)))  float  v8f;

#define B_    2
#define S_    2048
#define DM_   2048
#define H_    16
#define KV_   2
#define D_    128
#define G_    (H_ / KV_)
#define HALF_ (S_ / 2)
#define NQB_  (S_ / 32)
#define LDT   40   // padded leading dim (bf16 elems) for 32-wide LDS tiles

// ---------------------------------------------------------------------------
// CDNA5 async global->LDS staging (ASYNCcnt-tracked DMA path, no VGPR staging)
// lds_addr: 32-bit LDS byte address (flat->LDS truncates to addr[31:0]).
// ---------------------------------------------------------------------------
__device__ __forceinline__ void async_load_b128(unsigned lds_addr,
                                                const void* gptr) {
  asm volatile("global_load_async_to_lds_b128 %0, %1, off"
               :
               : "v"(lds_addr), "v"(gptr)
               : "memory");
}
__device__ __forceinline__ void wait_async0() {
  asm volatile("s_wait_asynccnt 0x0" ::: "memory");
}

// ---------------------------------------------------------------------------
// WMMA fragment loaders (wave32, 16x16x32 bf16)
// A 16x32 (MxK): lane m=lane&15; lanes 0-15 hold K=0..15, 16-31 hold
// K=16..31  (contiguous order, matching B)
// ---------------------------------------------------------------------------
__device__ __forceinline__ v16bf load_fragA(const bf16* p, int ld) {
  int lane = threadIdx.x & 31;
  int m  = lane & 15;
  int kh = (lane >> 4) << 4;                 // 0 or 16
  const bf16* r = p + (size_t)m * ld + kh;
  v8bf lo = *reinterpret_cast<const v8bf*>(r);        // K = kh .. kh+7
  v8bf hi = *reinterpret_cast<const v8bf*>(r + 8);    // K = kh+8 .. kh+15
  return __builtin_shufflevector(lo, hi, 0, 1, 2, 3, 4, 5, 6, 7,
                                 8, 9, 10, 11, 12, 13, 14, 15);
}

// B 32x16 (KxN), tile stored [n][k] row-major (i.e. W row-major): lane
// n=lane&15; lanes 0-15 hold K=0..15, lanes 16-31 hold K=16..31 (B striping)
__device__ __forceinline__ v16bf load_fragB_nmajor(const bf16* p, int ld) {
  int lane = threadIdx.x & 31;
  int n  = lane & 15;
  int kh = (lane >> 4) << 4;                 // 0 or 16
  const bf16* r = p + (size_t)n * ld + kh;
  v8bf lo = *reinterpret_cast<const v8bf*>(r);
  v8bf hi = *reinterpret_cast<const v8bf*>(r + 8);
  return __builtin_shufflevector(lo, hi, 0, 1, 2, 3, 4, 5, 6, 7,
                                 8, 9, 10, 11, 12, 13, 14, 15);
}

// B 32x16 (KxN), tile stored [k][n] row-major (e.g. V tile)
__device__ __forceinline__ v16bf load_fragB_kmajor(const bf16* p, int ld) {
  int lane = threadIdx.x & 31;
  int n  = lane & 15;
  int kh = (lane >> 4) << 4;
  v16bf f;
#pragma unroll
  for (int i = 0; i < 16; ++i) f[i] = p[(size_t)(kh + i) * ld + n];
  return f;
}

__device__ __forceinline__ v8f wmma_bf16(v16bf a, v16bf b, v8f c) {
  return __builtin_amdgcn_wmma_f32_16x16x32_bf16(
      false, a, false, b, (short)0, c, false, false);
}

// ---------------------------------------------------------------------------
// fp32 -> bf16 conversion
// ---------------------------------------------------------------------------
__global__ void f32_to_bf16_kernel(const float* __restrict__ in,
                                   bf16* __restrict__ out, int n) {
  int i = blockIdx.x * blockDim.x + threadIdx.x;
  if (i < n) out[i] = (bf16)in[i];
}

// ---------------------------------------------------------------------------
// Tiled GEMM: Y[M,N] = X[M,K] * W[N,K]^T (+bias). 128 threads = 4 waves,
// 64x64 workgroup tile, each wave computes 16 rows x 64 cols. Tiles staged
// via async global->LDS DMA.
// ---------------------------------------------------------------------------
template <bool OUT_BF16, bool HAS_BIAS>
__global__ __launch_bounds__(128) void gemm_xwt_kernel(
    const bf16* __restrict__ X, const bf16* __restrict__ W,
    const float* __restrict__ bias, bf16* __restrict__ Yb,
    float* __restrict__ Yf, int M, int N, int K) {
  __shared__ __align__(16) bf16 Xs[64 * LDT];
  __shared__ __align__(16) bf16 Ws[64 * LDT];
  const int n0 = blockIdx.x * 64;
  const int m0 = blockIdx.y * 64;
  const int tid = threadIdx.x;
  const int wave = tid >> 5;
  const int lane = tid & 31;
  const unsigned xsA = (unsigned)(size_t)(void*)Xs;
  const unsigned wsA = (unsigned)(size_t)(void*)Ws;

  v8f acc[4];
#pragma unroll
  for (int t = 0; t < 4; ++t)
#pragma unroll
    for (int e = 0; e < 8; ++e) acc[t][e] = 0.0f;

  for (int k0 = 0; k0 < K; k0 += 32) {
    __syncthreads();
    // 64 rows x 32 cols per tile = 256 x 16B chunks; 2 per thread, async DMA
#pragma unroll
    for (int i = 0; i < 2; ++i) {
      int u = tid + 128 * i;
      int row = u >> 2, c4 = u & 3;              // c4*8 bf16 = c4*16 bytes
      unsigned loff = (unsigned)((row * LDT + c4 * 8) * 2);
      async_load_b128(xsA + loff, &X[(size_t)(m0 + row) * K + k0 + c4 * 8]);
      async_load_b128(wsA + loff, &W[(size_t)(n0 + row) * K + k0 + c4 * 8]);
    }
    wait_async0();
    __syncthreads();
    v16bf a = load_fragA(&Xs[(wave * 16) * LDT], LDT);
#pragma unroll
    for (int nt = 0; nt < 4; ++nt) {
      v16bf b = load_fragB_nmajor(&Ws[(nt * 16) * LDT], LDT);
      acc[nt] = wmma_bf16(a, b, acc[nt]);
    }
  }

  const int half = lane >> 4;
  const int col0 = lane & 15;
#pragma unroll
  for (int nt = 0; nt < 4; ++nt) {
#pragma unroll
    for (int r = 0; r < 8; ++r) {
      int row = m0 + wave * 16 + r + 8 * half;
      int col = n0 + nt * 16 + col0;
      float v = acc[nt][r];
      if (HAS_BIAS) v += bias[col];
      if (OUT_BF16)
        Yb[(size_t)row * N + col] = (bf16)v;
      else
        Yf[(size_t)row * N + col] = v;
    }
  }
}

// ---------------------------------------------------------------------------
// RoPE + (B,S,NH*D) -> (B,NH,S,D) transpose. pos = s mod HALF_ (seq is split
// into two halves, each using cos/sin[0..HALF).
// ---------------------------------------------------------------------------
__global__ void rope_transpose_kernel(const bf16* __restrict__ T,
                                      const float* __restrict__ cs,
                                      const float* __restrict__ sn,
                                      bf16* __restrict__ Out, int NH) {
  int tid = blockIdx.x * blockDim.x + threadIdx.x;
  int total = B_ * NH * S_ * (D_ / 2);
  if (tid >= total) return;
  int d = tid & 63;
  int rest = tid >> 6;
  int s = rest % S_;
  int h = (rest / S_) % NH;
  int b = rest / (S_ * NH);
  int pos = (s < HALF_) ? s : s - HALF_;
  const float* cb = cs + ((size_t)b * HALF_ + pos) * D_;
  const float* sb = sn + ((size_t)b * HALF_ + pos) * D_;
  float c1 = cb[d], c2 = cb[d + 64];
  float s1 = sb[d], s2 = sb[d + 64];
  size_t in0 = ((size_t)(b * S_ + s) * NH + h) * D_;
  float x1 = (float)T[in0 + d];
  float x2 = (float)T[in0 + d + 64];
  size_t out0 = (((size_t)b * NH + h) * S_ + s) * D_;
  Out[out0 + d]      = (bf16)(x1 * c1 - x2 * s1);
  Out[out0 + d + 64] = (bf16)(x2 * c2 + x1 * s2);
}

__global__ void transpose_bhsd_kernel(const bf16* __restrict__ T,
                                      bf16* __restrict__ Out, int NH) {
  int tid = blockIdx.x * blockDim.x + threadIdx.x;
  int total = B_ * NH * S_ * D_;
  if (tid >= total) return;
  int d = tid & 127;
  int rest = tid >> 7;
  int s = rest % S_;
  int h = (rest / S_) % NH;
  int b = rest / (S_ * NH);
  Out[(((size_t)b * NH + h) * S_ + s) * D_ + d] =
      T[((size_t)(b * S_ + s) * NH + h) * D_ + d];
}

// ---------------------------------------------------------------------------
// Flash attention: one workgroup (64 thr = 2 waves) per (b, h, 32-row qblock).
// Block-causal mask (BLOCK=32) => key tile j is fully valid iff j <= qblk.
// Each wave owns a 16x128 Q fragment; K/V tiles staged with async DMA;
// online softmax; PV via WMMA. Output written as (B, S, H*D) bf16.
// ---------------------------------------------------------------------------
__global__ __launch_bounds__(64) void attn_fwd_kernel(
    const bf16* __restrict__ Q, const bf16* __restrict__ K,
    const bf16* __restrict__ V, bf16* __restrict__ O) {
  __shared__ __align__(16) bf16 Kt[32 * D_];
  __shared__ __align__(16) bf16 Vt[32 * D_];
  __shared__ __align__(16) bf16 Pt[2][16 * LDT];

  const int qb = blockIdx.x % NQB_;
  const int h  = (blockIdx.x / NQB_) % H_;
  const int b  = blockIdx.x / (NQB_ * H_);
  const int kvh = h / G_;
  const int tid = threadIdx.x;
  const int wave = tid >> 5;
  const int lane = tid & 31;
  const int half = lane >> 4;
  const int nidx = lane & 15;
  const unsigned ktA = (unsigned)(size_t)(void*)Kt;
  const unsigned vtA = (unsigned)(size_t)(void*)Vt;

  const bf16* qbase =
      Q + (((size_t)b * H_ + h) * S_ + (size_t)qb * 32 + wave * 16) * D_;
  const bf16* kbase = K + (((size_t)b * KV_ + kvh) * S_) * D_;
  const bf16* vbase = V + (((size_t)b * KV_ + kvh) * S_) * D_;

  v16bf qf[4];
#pragma unroll
  for (int kc = 0; kc < 4; ++kc) qf[kc] = load_fragA(qbase + 32 * kc, D_);

  const float scale = 0.08838834764831845f;  // 1/sqrt(128)
  float mrow[8], lrow[8];
  v8f oacc[8];
#pragma unroll
  for (int r = 0; r < 8; ++r) { mrow[r] = -3.0e38f; lrow[r] = 0.0f; }
#pragma unroll
  for (int t = 0; t < 8; ++t)
#pragma unroll
    for (int e = 0; e < 8; ++e) oacc[t][e] = 0.0f;

  for (int j = 0; j <= qb; ++j) {
    __syncthreads();
    {  // stage K/V tiles: 32x128 bf16 = 512 x 16B chunks, 8 per thread, async
      const char* kg = (const char*)(kbase + (size_t)j * 32 * D_);
      const char* vg = (const char*)(vbase + (size_t)j * 32 * D_);
#pragma unroll
      for (int i = 0; i < 8; ++i) {
        unsigned off = (unsigned)((tid + 64 * i) * 16);
        async_load_b128(ktA + off, kg + off);
        async_load_b128(vtA + off, vg + off);
      }
      wait_async0();
    }
    __syncthreads();

    // S = Q * K^T  (two 16-col key tiles)
    v8f c0, c1;
#pragma unroll
    for (int e = 0; e < 8; ++e) { c0[e] = 0.0f; c1[e] = 0.0f; }
#pragma unroll
    for (int kc = 0; kc < 4; ++kc) {
      v16bf b0 = load_fragB_nmajor(&Kt[0 * D_ + 32 * kc], D_);
      v16bf b1 = load_fragB_nmajor(&Kt[16 * D_ + 32 * kc], D_);
      c0 = wmma_bf16(qf[kc], b0, c0);
      c1 = wmma_bf16(qf[kc], b1, c1);
    }

    // online softmax over the 32 keys of this tile
    float p0[8], p1[8];
#pragma unroll
    for (int r = 0; r < 8; ++r) {
      float s0 = c0[r] * scale, s1 = c1[r] * scale;
      float mx = fmaxf(s0, s1);
#pragma unroll
      for (int o = 8; o >= 1; o >>= 1) mx = fmaxf(mx, __shfl_xor(mx, o, 32));
      float mnew = fmaxf(mrow[r], mx);
      float corr = __expf(mrow[r] - mnew);
      mrow[r] = mnew;
      float e0 = __expf(s0 - mnew);
      float e1 = __expf(s1 - mnew);
      float sum = e0 + e1;
#pragma unroll
      for (int o = 8; o >= 1; o >>= 1) sum += __shfl_xor(sum, o, 32);
      lrow[r] = lrow[r] * corr + sum;
#pragma unroll
      for (int t = 0; t < 8; ++t) oacc[t][r] *= corr;
      p0[r] = e0;
      p1[r] = e1;
    }

    // stage P (16x32 bf16) into LDS, re-read as an A fragment
#pragma unroll
    for (int r = 0; r < 8; ++r) {
      int m = r + 8 * half;
      Pt[wave][m * LDT + nidx]      = (bf16)p0[r];
      Pt[wave][m * LDT + nidx + 16] = (bf16)p1[r];
    }
    __syncthreads();
    v16bf pa = load_fragA(&Pt[wave][0], LDT);
#pragma unroll
    for (int nt = 0; nt < 8; ++nt) {
      v16bf vb = load_fragB_kmajor(&Vt[nt * 16], D_);
      oacc[nt] = wmma_bf16(pa, vb, oacc[nt]);
    }
  }

  // normalize and write O as (B, S, H*D)
  float inv[8];
#pragma unroll
  for (int r = 0; r < 8; ++r) inv[r] = 1.0f / lrow[r];
  size_t orow0 =
      ((size_t)b * S_ + (size_t)qb * 32 + wave * 16) * (H_ * D_) + (size_t)h * D_;
#pragma unroll
  for (int nt = 0; nt < 8; ++nt) {
#pragma unroll
    for (int r = 0; r < 8; ++r) {
      int m = r + 8 * half;
      O[orow0 + (size_t)m * (H_ * D_) + nt * 16 + nidx] =
          (bf16)(oacc[nt][r] * inv[r]);
    }
  }
}

// ---------------------------------------------------------------------------
// launcher
// ---------------------------------------------------------------------------
extern "C" void kernel_launch(void* const* d_in, const int* in_sizes, int n_in,
                              void* d_out, int out_size, void* d_ws,
                              size_t ws_size, hipStream_t stream) {
  (void)in_sizes; (void)n_in; (void)out_size; (void)ws_size;
  const float* hs = (const float*)d_in[0];
  const float* cs = (const float*)d_in[1];
  const float* sn = (const float*)d_in[2];
  // d_in[3] attention_mask: block-causal, derived analytically in-kernel
  const float* Wq = (const float*)d_in[4];
  const float* bq = (const float*)d_in[5];
  const float* Wk = (const float*)d_in[6];
  const float* bk = (const float*)d_in[7];
  const float* Wv = (const float*)d_in[8];
  const float* bv = (const float*)d_in[9];
  const float* Wo = (const float*)d_in[10];
  float* out = (float*)d_out;

  const int nHS = B_ * S_ * DM_;          // 8388608
  const int nWq = H_ * D_ * DM_;          // 4194304
  const int nWk = KV_ * D_ * DM_;         // 524288
  const int nQ  = B_ * S_ * H_ * D_;      // 8388608
  const int nK  = B_ * S_ * KV_ * D_;     // 1048576
  const int M   = B_ * S_;                // 4096

  char* ws = (char*)d_ws;
  size_t ofs = 0;
  auto alloc = [&](size_t bytes) {
    char* p = ws + ofs;
    ofs += (bytes + 255) & ~(size_t)255;
    return p;
  };
  bf16* hsb  = (bf16*)alloc((size_t)nHS * 2);
  bf16* wqb  = (bf16*)alloc((size_t)nWq * 2);
  bf16* wkb  = (bf16*)alloc((size_t)nWk * 2);
  bf16* wvb  = (bf16*)alloc((size_t)nWk * 2);
  bf16* wob  = (bf16*)alloc((size_t)nWq * 2);
  bf16* qtmp = (bf16*)alloc((size_t)nQ * 2);   // reused as attention output
  bf16* ktmp = (bf16*)alloc((size_t)nK * 2);
  bf16* vtmp = (bf16*)alloc((size_t)nK * 2);
  bf16* qT   = (bf16*)alloc((size_t)nQ * 2);
  bf16* kT   = (bf16*)alloc((size_t)nK * 2);
  bf16* vT   = (bf16*)alloc((size_t)nK * 2);
  bf16* obuf = qtmp;  // qtmp is dead after rope_q

  // 1) convert fp32 inputs to bf16
  f32_to_bf16_kernel<<<(nHS + 255) / 256, 256, 0, stream>>>(hs, hsb, nHS);
  f32_to_bf16_kernel<<<(nWq + 255) / 256, 256, 0, stream>>>(Wq, wqb, nWq);
  f32_to_bf16_kernel<<<(nWk + 255) / 256, 256, 0, stream>>>(Wk, wkb, nWk);
  f32_to_bf16_kernel<<<(nWk + 255) / 256, 256, 0, stream>>>(Wv, wvb, nWk);
  f32_to_bf16_kernel<<<(nWq + 255) / 256, 256, 0, stream>>>(Wo, wob, nWq);

  // 2) QKV projections (WMMA GEMM, fused bias)
  gemm_xwt_kernel<true, true><<<dim3((H_ * D_) / 64, M / 64), 128, 0, stream>>>(
      hsb, wqb, bq, qtmp, nullptr, M, H_ * D_, DM_);
  gemm_xwt_kernel<true, true><<<dim3((KV_ * D_) / 64, M / 64), 128, 0, stream>>>(
      hsb, wkb, bk, ktmp, nullptr, M, KV_ * D_, DM_);
  gemm_xwt_kernel<true, true><<<dim3((KV_ * D_) / 64, M / 64), 128, 0, stream>>>(
      hsb, wvb, bv, vtmp, nullptr, M, KV_ * D_, DM_);

  // 3) RoPE + transpose to (B, NH, S, D)
  {
    int tq = B_ * H_ * S_ * (D_ / 2);
    rope_transpose_kernel<<<(tq + 255) / 256, 256, 0, stream>>>(qtmp, cs, sn, qT, H_);
    int tk = B_ * KV_ * S_ * (D_ / 2);
    rope_transpose_kernel<<<(tk + 255) / 256, 256, 0, stream>>>(ktmp, cs, sn, kT, KV_);
    int tv = B_ * KV_ * S_ * D_;
    transpose_bhsd_kernel<<<(tv + 255) / 256, 256, 0, stream>>>(vtmp, vT, KV_);
  }

  // 4) flash attention with block-causal mask
  attn_fwd_kernel<<<B_ * H_ * NQB_, 64, 0, stream>>>(qT, kT, vT, obuf);

  // 5) output projection -> fp32 d_out
  gemm_xwt_kernel<false, false><<<dim3(DM_ / 64, M / 64), 128, 0, stream>>>(
      obuf, wob, nullptr, nullptr, out, M, DM_, H_ * D_);
}
